// Temporalatt_26594437496902
// MI455X (gfx1250) — matrix operand order repaired and checked by reading, not compiled, not run on hardware
//
#include <hip/hip_runtime.h>

typedef __bf16 bf16_t;
typedef __attribute__((ext_vector_type(16))) __bf16 v16bf;
typedef __attribute__((ext_vector_type(8)))  float  v8f;

#define B_ 8
#define T_ 512
#define N_ 12
#define D_ 1024
#define H_ 8
#define L_ 128
#define S_ (B_ * N_)   // 96

// ---------------- CDNA5 async global->LDS copy (ASYNCcnt path) ----------------
// Each lane copies 16 bytes: LDS[lds_off] = MEM[gaddr]. Tracked by ASYNCcnt.
__device__ __forceinline__ void async_b128(unsigned lds_off, const void* gaddr) {
    asm volatile("global_load_async_to_lds_b128 %0, %1, off"
                 :: "v"(lds_off), "v"(gaddr)
                 : "memory");
}
template <int N>
__device__ __forceinline__ void wait_asynccnt() {
    asm volatile("s_wait_asynccnt %0" :: "i"(N) : "memory");
}
// generic->LDS byte offset (addrspacecast AS3->generic puts the LDS offset in the low 32 bits)
__device__ __forceinline__ unsigned lds_off(const void* p) {
    return (unsigned)(unsigned long long)p;
}

// ---------------- WMMA fragment loaders (contiguous k order: element s of lane half h is k = 16h + s, same in A and B) ----------------

// A 16x32 bf16 fragment from row-major [M][K] tile (ldk = K stride).
__device__ __forceinline__ v16bf ldA(const bf16_t* p, int ldk, int lane) {
    const int row = lane & 15;
    const int kb  = (lane & 16) ? 16 : 0;
    const bf16_t* q = p + row * ldk + kb;
    v16bf a;
#pragma unroll
    for (int i = 0; i < 8; ++i) { a[i] = q[i]; a[8 + i] = q[8 + i]; }
    return a;
}

// B 32x16 bf16 fragment from row-major [N][K] tile (NT GEMM).
__device__ __forceinline__ v16bf ldB_nk(const bf16_t* p, int ldk, int lane) {
    const int col = lane & 15;
    const int kh  = (lane & 16) ? 16 : 0;
    const bf16_t* q = p + col * ldk + kh;
    v16bf b;
#pragma unroll
    for (int i = 0; i < 16; ++i) b[i] = q[i];
    return b;
}

// B 32x16 bf16 fragment from row-major [K][N] tile (NN GEMM).
__device__ __forceinline__ v16bf ldB_kn(const bf16_t* p, int ldn, int lane) {
    const int col = lane & 15;
    const int kh  = (lane & 16) ? 16 : 0;
    const bf16_t* q = p + kh * ldn + col;
    v16bf b;
#pragma unroll
    for (int i = 0; i < 16; ++i) b[i] = q[i * ldn];
    return b;
}

// ---------------- conversion kernels ----------------

// features [B,T,N,D] f32 -> xbf [S=b*N+n, T, D] bf16 (fused transpose)
__global__ __launch_bounds__(256) void cvt_x(const float* __restrict__ feat,
                                             bf16_t* __restrict__ xbf) {
    long i = (long)blockIdx.x * 256 + threadIdx.x;     // index into [S,T,D]
    int d  = (int)(i & (D_ - 1));
    long st = i >> 10;                                  // /D_
    int t = (int)(st % T_);
    int s = (int)(st / T_);
    int b = s / N_, n = s % N_;
    xbf[i] = (bf16_t)feat[(((long)b * T_ + t) * N_ + n) * D_ + d];
}

__global__ __launch_bounds__(256) void cvt_w(const float* __restrict__ src,
                                             bf16_t* __restrict__ dst, long n) {
    long i = (long)blockIdx.x * 256 + threadIdx.x;
    if (i < n) dst[i] = (bf16_t)src[i];
}

// ---------------- generic batched NT GEMM: C = A[M,K] x W[N,K]^T ----------------
// block tile 128x128, K-step 32; 8 waves as 2(M) x 4(N); 4x2 fragments per wave.
// Double-buffered LDS tiles filled with async global->LDS DMA; copy of tile i+1
// overlaps WMMA compute of tile i (gated by s_wait_asynccnt 4).
// mode 0: store bf16   mode 1: store f32
// mode 2: bf16 relu(x + bias)   mode 3: f32 (x + bias) scattered to [B,T,N,D]
__global__ __launch_bounds__(256) void gemm_bf16_nt(
    const bf16_t* __restrict__ A, const bf16_t* __restrict__ W,
    bf16_t* __restrict__ Cbf, float* __restrict__ Cf,
    const float* __restrict__ bias, float* __restrict__ outScat,
    int K, int lda, int ldb, int ldc,
    long strideA, long strideC, int mode) {
    __shared__ bf16_t sA[2][128 * 32];
    __shared__ bf16_t sB[2][128 * 32];

    const int tid = threadIdx.x;
    const int lane = tid & 31;
    const int wid = tid >> 5;
    const int wm = wid & 1, wn = wid >> 1;
    const int m0 = blockIdx.y * 128;
    const int n0 = blockIdx.x * 128;
    const long zA = (long)blockIdx.z * strideA;
    const long zC = (long)blockIdx.z * strideC;

    v8f acc[4][2] = {};

    // stage one 128x32 A tile + 128x32 B tile via async DMA (4 async ops/lane)
    auto stage = [&](int buf, int k0) {
#pragma unroll
        for (int it = 0; it < 2; ++it) {
            int c = tid + it * 256;            // 0..511
            int row = c >> 2, seg = (c & 3) * 8;
            async_b128(lds_off(&sA[buf][row * 32 + seg]),
                       &A[zA + (long)(m0 + row) * lda + k0 + seg]);
            async_b128(lds_off(&sB[buf][row * 32 + seg]),
                       &W[(long)(n0 + row) * ldb + k0 + seg]);
        }
    };

    const int ntiles = K >> 5;
    stage(0, 0);
    for (int i = 0; i < ntiles; ++i) {
        if (i + 1 < ntiles) {
            stage((i + 1) & 1, (i + 1) * 32);  // prefetch next tile
            wait_asynccnt<4>();                // current tile done; next in flight
        } else {
            wait_asynccnt<0>();
        }
        __syncthreads();

        const bf16_t* cA = sA[i & 1];
        const bf16_t* cB = sB[i & 1];
        v16bf af[4], bf2[2];
#pragma unroll
        for (int q = 0; q < 4; ++q) af[q] = ldA(&cA[(wm * 64 + q * 16) * 32], 32, lane);
#pragma unroll
        for (int j = 0; j < 2; ++j) bf2[j] = ldB_nk(&cB[(wn * 32 + j * 16) * 32], 32, lane);
#pragma unroll
        for (int q = 0; q < 4; ++q)
#pragma unroll
            for (int j = 0; j < 2; ++j)
                acc[q][j] = __builtin_amdgcn_wmma_f32_16x16x32_bf16(
                    false, af[q], false, bf2[j], (short)0, acc[q][j], false, false);
        __syncthreads();                        // readers done before buf reuse
    }

    const int ncolL = lane & 15;
    const int mb = (lane & 16) ? 8 : 0;
#pragma unroll
    for (int i = 0; i < 4; ++i) {
        const int mrow = m0 + wm * 64 + i * 16 + mb;
#pragma unroll
        for (int j = 0; j < 2; ++j) {
            const int ncol = n0 + wn * 32 + j * 16 + ncolL;
            const float bv = bias ? bias[ncol] : 0.f;
#pragma unroll
            for (int r = 0; r < 8; ++r) {
                float v = acc[i][j][r];
                const int mm = mrow + r;             // row (=t within batch)
                if (mode == 0) {
                    Cbf[zC + (long)mm * ldc + ncol] = (bf16_t)v;
                } else if (mode == 1) {
                    Cf[zC + (long)mm * ldc + ncol] = v;
                } else if (mode == 2) {
                    v += bv; v = v > 0.f ? v : 0.f;
                    Cbf[zC + (long)mm * ldc + ncol] = (bf16_t)v;
                } else {                              // scatter to [B,T,N,D]
                    v += bv;
                    const int s = blockIdx.z, bb = s / N_, nn = s % N_;
                    outScat[(((long)bb * T_ + mm) * N_ + nn) * D_ + ncol] = v;
                }
            }
        }
    }
}

// ---------------- fused bilinear attention: ax = (th ph^T / sqrt(D)) gx ----------------
// grid: (T/128, S*H); 128 rows of output per block; u-loop over 4 key tiles.
extern __shared__ char smem_raw[];
__global__ __launch_bounds__(256) void attn_fused(
    const bf16_t* __restrict__ th, const bf16_t* __restrict__ ph,
    const bf16_t* __restrict__ gx, bf16_t* __restrict__ ax) {
    bf16_t* sTh  = (bf16_t*)smem_raw;        // 128x128
    bf16_t* sPh  = sTh + 128 * 128;
    bf16_t* sGx  = sPh + 128 * 128;
    bf16_t* sAtt = sGx + 128 * 128;

    const int tid = threadIdx.x;
    const int lane = tid & 31;
    const int wid = tid >> 5;
    const int wm = wid & 1, wn = wid >> 1;
    const int t0 = blockIdx.x * 128;
    const int sh = blockIdx.y;
    const int s = sh >> 3, h = sh & 7;
    const int ld = H_ * L_;                               // 1024
    const long base = ((long)s * T_) * ld + (long)h * L_; // [s, 0, h*L]
    const bf16_t* thp = th + base;
    const bf16_t* php = ph + base;
    const bf16_t* gxp = gx + base;

    // stage query tile th[t0..t0+127][0..127] (async DMA, waited on in iter 0)
#pragma unroll
    for (int it = 0; it < 8; ++it) {
        int c = tid + it * 256;                // 0..2047
        int row = c >> 4, seg = (c & 15) * 8;
        async_b128(lds_off(&sTh[row * 128 + seg]),
                   &thp[(long)(t0 + row) * ld + seg]);
    }

    v8f accax[4][2] = {};
    const int ncolL = lane & 15;
    const int mb = (lane & 16) ? 8 : 0;

    for (int u0 = 0; u0 < T_; u0 += 128) {
        __syncthreads();                       // prev iter consumers done
#pragma unroll
        for (int it = 0; it < 8; ++it) {
            int c = tid + it * 256;
            int row = c >> 4, seg = (c & 15) * 8;
            async_b128(lds_off(&sPh[row * 128 + seg]),
                       &php[(long)(u0 + row) * ld + seg]);
            async_b128(lds_off(&sGx[row * 128 + seg]),
                       &gxp[(long)(u0 + row) * ld + seg]);
        }
        wait_asynccnt<0>();
        __syncthreads();

        // GEMM1: att[t,u] = sum_l th[t,l] * ph[u,l]   (NT, K=L=128)
        v8f acca[4][2] = {};
#pragma unroll
        for (int k0 = 0; k0 < 128; k0 += 32) {
            v16bf af[4], bf2[2];
#pragma unroll
            for (int i = 0; i < 4; ++i) af[i] = ldA(&sTh[(wm * 64 + i * 16) * 128 + k0], 128, lane);
#pragma unroll
            for (int j = 0; j < 2; ++j) bf2[j] = ldB_nk(&sPh[(wn * 32 + j * 16) * 128 + k0], 128, lane);
#pragma unroll
            for (int i = 0; i < 4; ++i)
#pragma unroll
                for (int j = 0; j < 2; ++j)
                    acca[i][j] = __builtin_amdgcn_wmma_f32_16x16x32_bf16(
                        false, af[i], false, bf2[j], (short)0, acca[i][j], false, false);
        }
        // scale by 1/sqrt(D)=1/32, stash as bf16 in LDS
#pragma unroll
        for (int i = 0; i < 4; ++i)
#pragma unroll
            for (int j = 0; j < 2; ++j)
#pragma unroll
                for (int r = 0; r < 8; ++r)
                    sAtt[(wm * 64 + i * 16 + mb + r) * 128 + wn * 32 + j * 16 + ncolL] =
                        (bf16_t)(acca[i][j][r] * 0.03125f);
        __syncthreads();

        // GEMM2: ax[t,l] += sum_u att[t,u] * gx[u,l]   (NN, K=128)
#pragma unroll
        for (int k0 = 0; k0 < 128; k0 += 32) {
            v16bf af[4], bf2[2];
#pragma unroll
            for (int i = 0; i < 4; ++i) af[i] = ldA(&sAtt[(wm * 64 + i * 16) * 128 + k0], 128, lane);
#pragma unroll
            for (int j = 0; j < 2; ++j) bf2[j] = ldB_kn(&sGx[k0 * 128 + wn * 32 + j * 16], 128, lane);
#pragma unroll
            for (int i = 0; i < 4; ++i)
#pragma unroll
                for (int j = 0; j < 2; ++j)
                    accax[i][j] = __builtin_amdgcn_wmma_f32_16x16x32_bf16(
                        false, af[i], false, bf2[j], (short)0, accax[i][j], false, false);
        }
    }

    // epilogue: multi[s, t, h*L + l] = ax  (bf16)
    bf16_t* axp = ax + base;
#pragma unroll
    for (int i = 0; i < 4; ++i)
#pragma unroll
        for (int j = 0; j < 2; ++j)
#pragma unroll
            for (int r = 0; r < 8; ++r)
                axp[(long)(t0 + wm * 64 + i * 16 + mb + r) * ld + wn * 32 + j * 16 + ncolL] =
                    (bf16_t)accax[i][j][r];
}

// ---------------- residual + layernorm ----------------
__global__ __launch_bounds__(256) void ln_residual(
    const float* __restrict__ wout, const float* __restrict__ feat,
    const float* __restrict__ g, const float* __restrict__ bb,
    bf16_t* __restrict__ out) {
    __shared__ float r1[256];
    __shared__ float r2[256];
    const int row = blockIdx.x;               // 0..S*T-1
    const int s = row / T_, t = row % T_;
    const int b = s / N_, n = s % N_;
    const float* xp = feat + (((long)b * T_ + t) * N_ + n) * D_;
    const float* wp = wout + (long)row * D_;

    float v[4], lsum = 0.f, lsq = 0.f;
#pragma unroll
    for (int i = 0; i < 4; ++i) {
        int c = threadIdx.x + i * 256;
        v[i] = wp[c] + xp[c];
        lsum += v[i]; lsq += v[i] * v[i];
    }
    r1[threadIdx.x] = lsum; r2[threadIdx.x] = lsq;
    __syncthreads();
    for (int off = 128; off > 0; off >>= 1) {
        if (threadIdx.x < off) {
            r1[threadIdx.x] += r1[threadIdx.x + off];
            r2[threadIdx.x] += r2[threadIdx.x + off];
        }
        __syncthreads();
    }
    const float mu  = r1[0] * (1.0f / D_);
    const float var = r2[0] * (1.0f / D_) - mu * mu;
    const float inv = rsqrtf(var + 1e-5f);
    bf16_t* op = out + (long)row * D_;
#pragma unroll
    for (int i = 0; i < 4; ++i) {
        int c = threadIdx.x + i * 256;
        op[c] = (bf16_t)((v[i] - mu) * inv * g[c] + bb[c]);
    }
}

// ---------------- host ----------------
extern "C" void kernel_launch(void* const* d_in, const int* in_sizes, int n_in,
                              void* d_out, int out_size, void* d_ws, size_t ws_size,
                              hipStream_t stream) {
    const float* feat  = (const float*)d_in[0];
    const float* thw_f = (const float*)d_in[1];
    const float* phw_f = (const float*)d_in[2];
    const float* gw_f  = (const float*)d_in[3];
    const float* wo_f  = (const float*)d_in[4];
    const float* lng   = (const float*)d_in[5];
    const float* lnb   = (const float*)d_in[6];
    const float* f1w_f = (const float*)d_in[7];
    const float* f1b   = (const float*)d_in[8];
    const float* f2w_f = (const float*)d_in[9];
    const float* f2b   = (const float*)d_in[10];
    float* out = (float*)d_out;

    char* ws = (char*)d_ws;
    const size_t SZX = (size_t)S_ * T_ * D_ * 2;   // one bf16 [S,T,D] tensor
    bf16_t* xbf   = (bf16_t*)(ws);
    bf16_t* thbf  = (bf16_t*)(ws + SZX);
    bf16_t* phbf  = (bf16_t*)(ws + 2 * SZX);
    bf16_t* gxbf  = (bf16_t*)(ws + 3 * SZX);
    bf16_t* axbf  = (bf16_t*)(ws + 4 * SZX);
    float*  woutf = (float*)(ws + 5 * SZX);        // f32 [S,T,D] = 2*SZX bytes
    bf16_t* wbf   = (bf16_t*)(ws + 7 * SZX);       // 6x 1M-element bf16 weights
    const long WN = 1024L * 1024L;
    bf16_t* thw = wbf,      *phw = wbf + WN,     *gww = wbf + 2 * WN;
    bf16_t* wow = wbf + 3 * WN, *f1w = wbf + 4 * WN, *f2w = wbf + 5 * WN;
    bf16_t* lnbf   = thbf;   // reuse (th dead after attention)
    bf16_t* relubf = phbf;   // reuse

    const long NX = (long)S_ * T_ * D_;
    cvt_x<<<(int)(NX / 256), 256, 0, stream>>>(feat, xbf);
    cvt_w<<<(int)(WN / 256), 256, 0, stream>>>(thw_f, thw, WN);
    cvt_w<<<(int)(WN / 256), 256, 0, stream>>>(phw_f, phw, WN);
    cvt_w<<<(int)(WN / 256), 256, 0, stream>>>(gw_f,  gww, WN);
    cvt_w<<<(int)(WN / 256), 256, 0, stream>>>(wo_f,  wow, WN);
    cvt_w<<<(int)(WN / 256), 256, 0, stream>>>(f1w_f, f1w, WN);
    cvt_w<<<(int)(WN / 256), 256, 0, stream>>>(f2w_f, f2w, WN);

    const dim3 gg(D_ / 128, T_ / 128, S_);          // (8,4,96)
    const long sTD = (long)T_ * D_;

    // projections (bf16 out): th/ph/gx[s,t,h*L+l]
    gemm_bf16_nt<<<gg, 256, 0, stream>>>(xbf, thw, thbf, nullptr, nullptr, nullptr,
                                         D_, D_, D_, D_, sTD, sTD, 0);
    gemm_bf16_nt<<<gg, 256, 0, stream>>>(xbf, phw, phbf, nullptr, nullptr, nullptr,
                                         D_, D_, D_, D_, sTD, sTD, 0);
    gemm_bf16_nt<<<gg, 256, 0, stream>>>(xbf, gww, gxbf, nullptr, nullptr, nullptr,
                                         D_, D_, D_, D_, sTD, sTD, 0);

    // fused attention -> multi (bf16), 128KB dynamic LDS
    attn_fused<<<dim3(T_ / 128, S_ * H_), 256, 4 * 128 * 128 * 2, stream>>>(
        thbf, phbf, gxbf, axbf);

    // wout projection (f32 out for LN precision)
    gemm_bf16_nt<<<gg, 256, 0, stream>>>(axbf, wow, nullptr, woutf, nullptr, nullptr,
                                         D_, D_, D_, D_, sTD, sTD, 1);
    // residual + layernorm -> bf16
    ln_residual<<<S_ * T_, 256, 0, stream>>>(woutf, feat, lng, lnb, lnbf);
    // ff1 + bias + relu (bf16 out)
    gemm_bf16_nt<<<gg, 256, 0, stream>>>(lnbf, f1w, relubf, nullptr, f1b, nullptr,
                                         D_, D_, D_, D_, sTD, sTD, 2);
    // ff2 + bias, scattered back to [B,T,N,D] f32 output
    gemm_bf16_nt<<<gg, 256, 0, stream>>>(relubf, f2w, nullptr, nullptr, f2b, out,
                                         D_, D_, D_, D_, sTD, sTD, 3);
}
